// WanSelfAttention_28922309771780
// MI455X (gfx1250) — compile-verified, matrix operand order checked by reading, not run on hardware
//
#include <hip/hip_runtime.h>
#include <hip/hip_bf16.h>

#define DIM   1536
#define HEADS 12
#define HD    128   // head dim
// RoPE split of HD/2 = 64 rotary pairs: c_f=22 (frame), c_h=21, c_w=21

typedef __bf16 bf16_t;
typedef __attribute__((ext_vector_type(16))) __bf16 v16bf;
typedef __attribute__((ext_vector_type(8)))  __bf16 v8bf;
typedef __attribute__((ext_vector_type(8)))  float   v8f;

// ---------------------------------------------------------------------------
// WMMA fragment helpers (wave32, V_WMMA_F32_16X16X32_BF16).
// 16-bit A-matrix 16x32 layout (ISA 7.12.2): lane<16 holds row M=lane with
// K 0-7 (v0..3) and K 16-23 (v4..7); lane>=16 holds row M=lane-16 with
// K 8-15 and K 24-31.  B is supplied as B^T [N][K] with the same pattern
// (lane -> N).  So one loader serves both operands from row-major storage.
// ---------------------------------------------------------------------------
__device__ __forceinline__ v16bf frag_from_ptr(const bf16_t* p)
{
    v8bf a = *(const v8bf*)(p);        // K = k0 + 8*half     .. +7
    v8bf b = *(const v8bf*)(p + 16);   // K = k0 + 8*half + 16.. +7
    return __builtin_shufflevector(a, b, 0, 1, 2, 3, 4, 5, 6, 7,
                                         8, 9, 10, 11, 12, 13, 14, 15);
}

__device__ __forceinline__ v16bf load_frag_rowmajor(const bf16_t* __restrict__ base,
                                                    int row0, int k0, int ld)
{
    const int lane = threadIdx.x & 31;
    const int lo   = lane & 15;
    const int half = lane >> 4;
    return frag_from_ptr(base + (size_t)(row0 + lo) * (size_t)ld + (size_t)(k0 + 8 * half));
}

__device__ __forceinline__ v8f wmma_bf16(v16bf a, v16bf b, v8f c)
{
    // (neg_a, A, neg_b, B, c_mod, C, reuse_a, reuse_b)
    return __builtin_amdgcn_wmma_f32_16x16x32_bf16(false, a, false, b, (short)0, c,
                                                   false, false);
}

// ---------------------------------------------------------------------------
// CDNA5 async global->LDS DMA (per-lane 16B), tracked by ASYNCcnt.
// ---------------------------------------------------------------------------
__device__ __forceinline__ void async_load_b128(bf16_t* lds_ptr, const bf16_t* gptr)
{
    unsigned loff = (unsigned)(unsigned long long)(uintptr_t)lds_ptr;  // LDS offset = addr[31:0]
    unsigned long long ga = (unsigned long long)(uintptr_t)gptr;
    asm volatile("global_load_async_to_lds_b128 %0, %1, off"
                 :: "v"(loff), "v"(ga) : "memory");
}

__device__ __forceinline__ void wait_asynccnt0()
{
    asm volatile("s_wait_asynccnt 0x0" ::: "memory");
}

// ---------------------------------------------------------------------------
// fp32 -> bf16 elementwise convert
// ---------------------------------------------------------------------------
__global__ void k_f32_to_bf16(const float* __restrict__ src, bf16_t* __restrict__ dst,
                              long n)
{
    long i = (long)blockIdx.x * blockDim.x + threadIdx.x;
    if (i < n) dst[i] = (bf16_t)src[i];
}

// ---------------------------------------------------------------------------
// W [K][N] f32  ->  Wt [N][K] bf16   (LDS 32x32 tile transpose)
// ---------------------------------------------------------------------------
__global__ void k_transpose_to_bf16(const float* __restrict__ W, bf16_t* __restrict__ Wt,
                                    int K, int N)
{
    __shared__ float tile[32][33];
    const int n0 = blockIdx.x * 32;
    const int k0 = blockIdx.y * 32;
    const int tx = threadIdx.x;   // 0..31
    const int ty = threadIdx.y;   // 0..7
#pragma unroll
    for (int i = 0; i < 32; i += 8)
        tile[ty + i][tx] = W[(size_t)(k0 + ty + i) * N + (n0 + tx)];
    __syncthreads();
#pragma unroll
    for (int i = 0; i < 32; i += 8)
        Wt[(size_t)(n0 + ty + i) * K + (k0 + tx)] = (bf16_t)tile[tx][ty + i];
}

// ---------------------------------------------------------------------------
// WMMA GEMM:  C[M][N] = A[M][K] * Bt[N][K]^T + bias
//   MODE 0: store f32 row-major into Cf
//   MODE 1: store bf16 TRANSPOSED into Cb[N][M]   (used for V^T per head)
// Block = 128 threads = 4 waves; wave tile 64x64, block tile 128x128.
// ---------------------------------------------------------------------------
template <int MODE>
__global__ void __launch_bounds__(128, 1)
k_gemm_wmma(const bf16_t* __restrict__ A,
            const bf16_t* __restrict__ Bt,
            const float*  __restrict__ bias,
            float*  __restrict__ Cf,
            bf16_t* __restrict__ Cb,
            int M, int N, int K)
{
    const int lane  = threadIdx.x & 31;
    const int lo    = lane & 15;
    const int half  = lane >> 4;
    const int wave  = threadIdx.x >> 5;                    // 0..3
    const int mbase = blockIdx.y * 128 + (wave >> 1) * 64;
    const int nbase = blockIdx.x * 128 + (wave & 1) * 64;

    v8f c[4][4];
#pragma unroll
    for (int i = 0; i < 4; ++i)
#pragma unroll
        for (int j = 0; j < 4; ++j)
#pragma unroll
            for (int r = 0; r < 8; ++r) c[i][j][r] = 0.f;

    for (int k0 = 0; k0 < K; k0 += 32) {
        v16bf a[4], b[4];
#pragma unroll
        for (int i = 0; i < 4; ++i) a[i] = load_frag_rowmajor(A,  mbase + 16 * i, k0, K);
#pragma unroll
        for (int j = 0; j < 4; ++j) b[j] = load_frag_rowmajor(Bt, nbase + 16 * j, k0, K);
#pragma unroll
        for (int i = 0; i < 4; ++i)
#pragma unroll
            for (int j = 0; j < 4; ++j)
                c[i][j] = wmma_bf16(a[i], b[j], c[i][j]);
    }

    // C/D layout: lane lo = column n, VGPR r = row (m = r + 8*half)
#pragma unroll
    for (int i = 0; i < 4; ++i) {
#pragma unroll
        for (int j = 0; j < 4; ++j) {
            const int n  = nbase + 16 * j + lo;
            const float bv = bias[n];
            if (MODE == 0) {
#pragma unroll
                for (int r = 0; r < 8; ++r) {
                    const int m = mbase + 16 * i + 8 * half + r;
                    Cf[(size_t)m * N + n] = c[i][j][r] + bv;
                }
            } else {
                v8bf st;
#pragma unroll
                for (int r = 0; r < 8; ++r) st[r] = (bf16_t)(c[i][j][r] + bv);
                // contiguous run of 8 rows m -> 16B store into Cb[n][m]
                *(v8bf*)(Cb + (size_t)n * M + (mbase + 16 * i + 8 * half)) = st;
            }
        }
    }
}

// ---------------------------------------------------------------------------
// Fused RMS-norm (over full DIM row) + RoPE -> bf16.  One block per row.
// ---------------------------------------------------------------------------
__global__ void k_rmsrope(const float* __restrict__ T,      // [L][DIM] f32
                          const float* __restrict__ g,      // [DIM]
                          const float* __restrict__ freqs,  // [1024][64]
                          const int*   __restrict__ grid_sizes, // [F,H,W]
                          bf16_t* __restrict__ out,         // [L][DIM] bf16
                          int L)
{
    const int l = blockIdx.x;
    const float* row = T + (size_t)l * DIM;
    __shared__ float red[8];

    float ss = 0.f;
    for (int i = threadIdx.x; i < DIM; i += blockDim.x) {
        float v = row[i];
        ss += v * v;
    }
#pragma unroll
    for (int off = 16; off; off >>= 1) ss += __shfl_xor(ss, off);
    if ((threadIdx.x & 31) == 0) red[threadIdx.x >> 5] = ss;
    __syncthreads();
    if (threadIdx.x < 8) {
        float v = red[threadIdx.x];
#pragma unroll
        for (int off = 4; off; off >>= 1) v += __shfl_xor(v, off);
        if (threadIdx.x == 0) red[0] = v;
    }
    __syncthreads();
    const float scale = rsqrtf(red[0] / (float)DIM + 1e-6f);

    const int Hg = grid_sizes[1], Wg = grid_sizes[2];
    const int w_idx = l % Wg;
    const int h_idx = (l / Wg) % Hg;
    const int f_idx = l / (Wg * Hg);

    const int c   = HD / 2;                 // 64 rotary pairs per head
    const int c_h = c / 3;                  // 21
    const int c_f = c - 2 * (c / 3);        // 22

    for (int p = threadIdx.x; p < HEADS * c; p += blockDim.x) {
        const int head = p / c;
        const int i    = p % c;
        int pos;
        if (i < c_f)            pos = f_idx;
        else if (i < c_f + c_h) pos = h_idx;
        else                    pos = w_idx;
        const float ang = freqs[(size_t)pos * c + i];
        const float cs = __cosf(ang);
        const float sn = __sinf(ang);
        const int col = head * HD + 2 * i;
        const float a = row[col]     * scale * g[col];
        const float b = row[col + 1] * scale * g[col + 1];
        out[(size_t)l * DIM + col]     = (bf16_t)(a * cs - b * sn);
        out[(size_t)l * DIM + col + 1] = (bf16_t)(a * sn + b * cs);
    }
}

// ---------------------------------------------------------------------------
// Flash attention.  Block = 4 waves, ALL on the SAME head (4 adjacent q-tiles
// of 16 rows), so the per-block 32-key K-tile and V^T-tile are staged once in
// LDS via global_load_async_to_lds_b128 (double-buffered, ASYNCcnt-tracked)
// and shared by all 4 waves: 4x less L2 traffic for K/V.
//   S^T = K . Q^T        (lane column = query; stats in-lane + one xor16)
//   O^T = V^T . P^T      (P^T's C-layout == B-fragment layout for K=32 keys)
// ---------------------------------------------------------------------------
#define KSTR 136   // K-tile LDS row stride in bf16 (128 data + 8 pad)
#define VSTR 40    // V^T-tile LDS row stride in bf16 (32 data + 8 pad)

__global__ void __launch_bounds__(128, 1)
k_attention(const bf16_t* __restrict__ Qb,   // [L][DIM]
            const bf16_t* __restrict__ Kb,   // [L][DIM]
            const bf16_t* __restrict__ Vt,   // [DIM][L]  (row = head*HD + d)
            const int*    __restrict__ seq_lens,
            bf16_t* __restrict__ Ob,         // [L][DIM]
            int L)
{
    __shared__ bf16_t skbuf[2][32 * KSTR];   // K tile: 32 keys x 128 d
    __shared__ bf16_t svbuf[2][128 * VSTR];  // V^T tile: 128 d x 32 keys

    const int qtiles = L >> 4;
    const int bph    = qtiles >> 2;            // blocks per head
    const int head   = blockIdx.x / bph;
    const int wave   = threadIdx.x >> 5;
    const int q0     = ((blockIdx.x % bph) * 4 + wave) << 4;
    const int hc     = head * HD;
    const int tid    = threadIdx.x;
    const int lane   = threadIdx.x & 31;
    const int lo     = lane & 15;
    const int half   = lane >> 4;
    const int seqlen = seq_lens[0];
    const float sm_scale = 0.08838834764831845f;   // HD^-0.5

    // cooperative async fill of one 32-key tile (K: 256 x 16B, V^T: 512 x 16B)
    auto issue_tile = [&](int kb, int buf) {
#pragma unroll
        for (int s = 0; s < 2; ++s) {              // K tile segments
            const int seg = tid + 128 * s;         // 0..255
            const int row = seg >> 3, c16 = seg & 7;
            async_load_b128(&skbuf[buf][row * KSTR + c16 * 16],
                            Kb + (size_t)(kb + row) * DIM + hc + c16 * 16);
        }
#pragma unroll
        for (int s = 0; s < 4; ++s) {              // V^T tile segments
            const int seg = tid + 128 * s;         // 0..511
            const int row = seg >> 2, c8 = (seg & 3) * 8;
            async_load_b128(&svbuf[buf][row * VSTR + c8],
                            Vt + (size_t)(hc + row) * L + kb + c8);
        }
    };

    // persistent Q as B^T fragments (4 x K=32 covering d=128)
    v16bf qf[4];
#pragma unroll
    for (int kk = 0; kk < 4; ++kk)
        qf[kk] = load_frag_rowmajor(Qb, q0, hc + 32 * kk, DIM);

    v8f o[8];                                      // O^T: 128 d-rows x 16 queries
#pragma unroll
    for (int j = 0; j < 8; ++j)
#pragma unroll
        for (int r = 0; r < 8; ++r) o[j][r] = 0.f;
    float m_i = -3.0e38f, l_i = 0.f;

    const int nkb = (seqlen + 31) / 32;
    issue_tile(0, 0);
    wait_asynccnt0();
    __syncthreads();

    for (int t = 0; t < nkb; ++t) {
        const int kb  = t * 32;
        const int buf = t & 1;
        if (t + 1 < nkb) issue_tile(kb + 32, buf ^ 1);

        const bf16_t* kt = &skbuf[buf][0];
        const bf16_t* vt = &svbuf[buf][0];

        v8f s0, s1;
#pragma unroll
        for (int r = 0; r < 8; ++r) { s0[r] = 0.f; s1[r] = 0.f; }
#pragma unroll
        for (int kk = 0; kk < 4; ++kk) {
            v16bf ka = frag_from_ptr(kt + (lo)      * KSTR + kk * 32 + 8 * half);
            s0 = wmma_bf16(ka, qf[kk], s0);        // keys kb..kb+15
        }
#pragma unroll
        for (int kk = 0; kk < 4; ++kk) {
            v16bf ka = frag_from_ptr(kt + (16 + lo) * KSTR + kk * 32 + 8 * half);
            s1 = wmma_bf16(ka, qf[kk], s1);        // keys kb+16..kb+31
        }

        float p0[8], p1[8];
        float bm = -3.0e38f;
#pragma unroll
        for (int r = 0; r < 8; ++r) {
            p0[r] = s0[r] * sm_scale;
            p1[r] = s1[r] * sm_scale;
            bm = fmaxf(bm, fmaxf(p0[r], p1[r]));
        }
        bm = fmaxf(bm, __shfl_xor(bm, 16));        // combine key halves
        const float m_new = fmaxf(m_i, bm);
        const float alpha = __expf(m_i - m_new);

        float rs = 0.f;
        v16bf pf;                                  // == B-fragment over 32 keys
#pragma unroll
        for (int r = 0; r < 8; ++r) {
            const int key0 = kb + 8 * half + r;
            const int key1 = key0 + 16;
            const float e0 = (key0 < seqlen) ? __expf(p0[r] - m_new) : 0.f;
            const float e1 = (key1 < seqlen) ? __expf(p1[r] - m_new) : 0.f;
            rs += e0 + e1;
            pf[r]     = (bf16_t)e0;
            pf[r + 8] = (bf16_t)e1;
        }
        rs += __shfl_xor(rs, 16);
        l_i = l_i * alpha + rs;
        m_i = m_new;

#pragma unroll
        for (int j = 0; j < 8; ++j)
#pragma unroll
            for (int r = 0; r < 8; ++r) o[j][r] *= alpha;

#pragma unroll
        for (int j = 0; j < 8; ++j) {
            v16bf va = frag_from_ptr(vt + (j * 16 + lo) * VSTR + 8 * half);
            o[j] = wmma_bf16(va, pf, o[j]);        // O^T += V^T . P^T
        }

        wait_asynccnt0();                          // next tile landed (per wave)
        __syncthreads();                           // all waves done with `buf`
    }

    const float inv = (l_i > 0.f) ? (1.f / l_i) : 0.f;
#pragma unroll
    for (int j = 0; j < 8; ++j) {
        v8bf st;
#pragma unroll
        for (int r = 0; r < 8; ++r) st[r] = (bf16_t)(o[j][r] * inv);
        // O^T lane = query q0+lo; rows d contiguous -> 16B store into Ob[q][hc+d]
        *(v8bf*)(Ob + (size_t)(q0 + lo) * DIM + hc + 16 * j + 8 * half) = st;
    }
}

// ---------------------------------------------------------------------------
// Host launcher
// ---------------------------------------------------------------------------
extern "C" void kernel_launch(void* const* d_in, const int* in_sizes, int n_in,
                              void* d_out, int out_size, void* d_ws, size_t ws_size,
                              hipStream_t stream)
{
    const float* x          = (const float*)d_in[0];
    const int*   seq_lens   = (const int*)  d_in[1];
    const int*   grid_sizes = (const int*)  d_in[2];
    const float* freqs      = (const float*)d_in[3];
    const float* Wq = (const float*)d_in[4];
    const float* bq = (const float*)d_in[5];
    const float* Wk = (const float*)d_in[6];
    const float* bk = (const float*)d_in[7];
    const float* Wv = (const float*)d_in[8];
    const float* bv = (const float*)d_in[9];
    const float* Wo = (const float*)d_in[10];
    const float* bo = (const float*)d_in[11];
    const float* gq = (const float*)d_in[12];
    const float* gk = (const float*)d_in[13];
    float* out = (float*)d_out;

    const int L = in_sizes[0] / DIM;   // 4096

    // --- workspace layout (256B aligned); ~94 MB total -----------------
    char*  ws  = (char*)d_ws;
    size_t off = 0;
    auto alloc = [&](size_t bytes) -> char* {
        char* p = ws + off;
        off += (bytes + 255) & ~(size_t)255;
        return p;
    };
    bf16_t* xb   = (bf16_t*)alloc((size_t)L * DIM * sizeof(bf16_t));
    bf16_t* Wqt  = (bf16_t*)alloc((size_t)DIM * DIM * sizeof(bf16_t));
    bf16_t* Wkt  = (bf16_t*)alloc((size_t)DIM * DIM * sizeof(bf16_t));
    bf16_t* Wvt  = (bf16_t*)alloc((size_t)DIM * DIM * sizeof(bf16_t));
    bf16_t* Wot  = (bf16_t*)alloc((size_t)DIM * DIM * sizeof(bf16_t));
    float*  F32S = (float*) alloc((size_t)L * DIM * sizeof(float)); // Q rows, then K rows
    bf16_t* Qbuf = (bf16_t*)alloc((size_t)L * DIM * sizeof(bf16_t));
    bf16_t* Kbuf = (bf16_t*)alloc((size_t)L * DIM * sizeof(bf16_t));
    bf16_t* Vt   = (bf16_t*)alloc((size_t)L * DIM * sizeof(bf16_t));
    bf16_t* Ob   = (bf16_t*)F32S;      // alias: f32 scratch is dead by attention time

    // 1) convert x to bf16
    {
        long n = (long)L * DIM;
        k_f32_to_bf16<<<(unsigned)((n + 255) / 256), 256, 0, stream>>>(x, xb, n);
    }
    // 2) transpose weights to bf16 [N][K]
    {
        dim3 tb(32, 8);
        dim3 tg(DIM / 32, DIM / 32);
        k_transpose_to_bf16<<<tg, tb, 0, stream>>>(Wq, Wqt, DIM, DIM);
        k_transpose_to_bf16<<<tg, tb, 0, stream>>>(Wk, Wkt, DIM, DIM);
        k_transpose_to_bf16<<<tg, tb, 0, stream>>>(Wv, Wvt, DIM, DIM);
        k_transpose_to_bf16<<<tg, tb, 0, stream>>>(Wo, Wot, DIM, DIM);
    }

    dim3 gg(DIM / 128, L / 128);   // (12, 32), 128 threads = 4 waves/block

    // 3) Q = x@Wq + bq (f32) -> RMS+RoPE -> bf16
    k_gemm_wmma<0><<<gg, 128, 0, stream>>>(xb, Wqt, bq, F32S, nullptr, L, DIM, DIM);
    k_rmsrope<<<L, 256, 0, stream>>>(F32S, gq, freqs, grid_sizes, Qbuf, L);
    // 4) K = x@Wk + bk (f32) -> RMS+RoPE -> bf16   (reuses f32 scratch)
    k_gemm_wmma<0><<<gg, 128, 0, stream>>>(xb, Wkt, bk, F32S, nullptr, L, DIM, DIM);
    k_rmsrope<<<L, 256, 0, stream>>>(F32S, gk, freqs, grid_sizes, Kbuf, L);
    // 5) V = x@Wv + bv, stored transposed per head: Vt[head*HD + d][l]
    k_gemm_wmma<1><<<gg, 128, 0, stream>>>(xb, Wvt, bv, nullptr, Vt, L, DIM, DIM);

    // 6) flash attention: block = 4 waves on one head sharing LDS K/V tiles
    {
        const int blocks = HEADS * ((L / 16) / 4);   // 768
        k_attention<<<blocks, 128, 0, stream>>>(Qbuf, Kbuf, Vt, seq_lens, Ob, L);
    }

    // 7) out = attn_out @ Wo + bo (f32 to d_out)
    k_gemm_wmma<0><<<gg, 128, 0, stream>>>(Ob, Wot, bo, out, nullptr, L, DIM, DIM);
}
